// SelfAttention_5823975653747
// MI455X (gfx1250) — matrix operand order repaired and checked by reading, not compiled, not run on hardware
//
#include <hip/hip_runtime.h>
#include <hip/hip_bf16.h>

typedef __attribute__((ext_vector_type(16))) _Float16 v16h;
typedef __attribute__((ext_vector_type(8)))  float    v8f;

#define BATCH 4
#define C_DIM 64
#define QKD   8
#define NPIX  4096
#define JSEG  (NPIX / 4)   // columns per wave (split-j)

union F16x16 { v16h v; uint4 u4[2]; _Float16 h[16]; };
union F32x8  { v8f  v; float f[8]; };

// ---------------------------------------------------------------------------
// DPP16 butterfly reductions across a 16-lane group (stays within each
// half-wave: DPP rows are lanes 0-15 / 16-31).
// ---------------------------------------------------------------------------
__device__ __forceinline__ float dpp_xchg(float x, const int ctrl_sel) {
  union { float f; int i; } a, b;
  a.f = x;
  switch (ctrl_sel) {
    case 0: b.i = __builtin_amdgcn_mov_dpp(a.i, 0xB1, 0xf, 0xf, true); break;   // quad_perm [1,0,3,2]
    case 1: b.i = __builtin_amdgcn_mov_dpp(a.i, 0x4E, 0xf, 0xf, true); break;   // quad_perm [2,3,0,1]
    case 2: b.i = __builtin_amdgcn_mov_dpp(a.i, 0x141, 0xf, 0xf, true); break;  // row_half_mirror
    default: b.i = __builtin_amdgcn_mov_dpp(a.i, 0x140, 0xf, 0xf, true); break; // row_mirror
  }
  return b.f;
}

__device__ __forceinline__ float red_max16(float x) {
  x = fmaxf(x, dpp_xchg(x, 0));
  x = fmaxf(x, dpp_xchg(x, 1));
  x = fmaxf(x, dpp_xchg(x, 2));
  x = fmaxf(x, dpp_xchg(x, 3));
  return x;
}

__device__ __forceinline__ float red_sum16(float x) {
  x += dpp_xchg(x, 0);
  x += dpp_xchg(x, 1);
  x += dpp_xchg(x, 2);
  x += dpp_xchg(x, 3);
  return x;
}

// ---------------------------------------------------------------------------
// Kernel 1: 1x1-conv projections, fp32 -> f16 Q/K/V
//   Qh[b][n][o], Kh[b][n][o] (8 dims contiguous), Vh[b][c][n]
// ---------------------------------------------------------------------------
__global__ void sa_proj_kernel(const float* __restrict__ x,
                               const float* __restrict__ Wq, const float* __restrict__ bq,
                               const float* __restrict__ Wk, const float* __restrict__ bk,
                               const float* __restrict__ Wv, const float* __restrict__ bv,
                               _Float16* __restrict__ Qh,
                               _Float16* __restrict__ Kh,
                               _Float16* __restrict__ Vh) {
  int gid = blockIdx.x * blockDim.x + threadIdx.x;     // [0, B*N)
  int b = gid >> 12;
  int n = gid & (NPIX - 1);

  const float* xb = x + ((size_t)b * C_DIM) * NPIX + n;
  float xv[C_DIM];
#pragma unroll
  for (int c = 0; c < C_DIM; ++c) xv[c] = xb[(size_t)c * NPIX];

  for (int o = 0; o < QKD; ++o) {
    float aq = bq[o];
    float ak = bk[o];
#pragma unroll
    for (int c = 0; c < C_DIM; ++c) {
      aq = fmaf(Wq[o * C_DIM + c], xv[c], aq);
      ak = fmaf(Wk[o * C_DIM + c], xv[c], ak);
    }
    Qh[((size_t)b * NPIX + n) * QKD + o] = (_Float16)aq;
    Kh[((size_t)b * NPIX + n) * QKD + o] = (_Float16)ak;
  }

  for (int o = 0; o < C_DIM; ++o) {
    float av = bv[o];
#pragma unroll
    for (int c = 0; c < C_DIM; ++c) av = fmaf(Wv[o * C_DIM + c], xv[c], av);
    Vh[((size_t)b * C_DIM + o) * NPIX + n] = (_Float16)av;
  }
}

// ---------------------------------------------------------------------------
// Kernel 2: split-j flash attention. Block = 16 query rows; each of the 4
// waves owns a disjoint 1024-column j segment with private (m, l, acc);
// partials merged through LDS at the end. Per 64-col tile: 4 S-WMMAs
// (K=8 zero-padded to 32), DPP online softmax, LDS D->A permute of P,
// 8 PV WMMAs.
// ---------------------------------------------------------------------------
__global__ void __launch_bounds__(128)
sa_attn_kernel(const _Float16* __restrict__ Qh,
               const _Float16* __restrict__ Kh,
               const _Float16* __restrict__ Vh,
               const float* __restrict__ x,
               const float* __restrict__ gamma,
               float* __restrict__ out) {
  const int lane = threadIdx.x & 31;
  const int wave = threadIdx.x >> 5;
  const int idx  = blockIdx.x;               // [0, 1024)
  const int b    = idx >> 8;
  const int i0   = (idx & 255) << 4;         // query row block start
  const int g    = lane >> 4;                // half-wave group
  const int ln   = lane & 15;
  const unsigned msk = (lane < 16) ? 0xFFFFFFFFu : 0u;  // zero for upper half
  const int jbase = wave * JSEG;

  // per-wave P staging (D->A layout permute) + merge buffers
  __shared__ __attribute__((aligned(16))) _Float16 Pst[4][16 * 64];
  __shared__ float accS[4][4][8 * 32];   // [wave][cfrag][r*32+lane]
  __shared__ float mS[4][8 * 32];        // [wave][r*32+lane]
  __shared__ float lS[4][8 * 32];
  _Float16* P = Pst[wave];

  const _Float16* Kb = Kh + (size_t)b * NPIX * QKD;
  const _Float16* Vb = Vh + (size_t)b * C_DIM * NPIX;

  // ---- Q A-fragment (loop-invariant) ----
  // A layout: M = lane%16; half h -> K = (h>=8?16:0) + h%8 + 8*(lane/16).
  // Valid K = 0..7 => lanes 0-15, halves 0..7; rest zero (branch-free mask).
  F16x16 qa;
  {
    uint4 t = *(const uint4*)(Qh + ((size_t)b * NPIX + i0 + ln) * QKD);
    qa.u4[0] = make_uint4(t.x & msk, t.y & msk, t.z & msk, t.w & msk);
    qa.u4[1] = make_uint4(0u, 0u, 0u, 0u);
  }

  v8f zf = {};
  F32x8 acc[4];
#pragma unroll
  for (int f = 0; f < 4; ++f) acc[f].v = zf;

  float mrow[8], lrow[8];
#pragma unroll
  for (int r = 0; r < 8; ++r) { mrow[r] = -1e30f; lrow[r] = 0.0f; }

  for (int j0 = jbase; j0 < jbase + JSEG; j0 += 64) {
    // ---- S = Q K^T over 4 subtiles of 16 columns (branch-free loads) ----
    F32x8 s[4];
#pragma unroll
    for (int st = 0; st < 4; ++st) {
      F16x16 kb;
      uint4 t = *(const uint4*)(Kb + (size_t)(j0 + st * 16 + ln) * QKD);
      kb.u4[0] = make_uint4(t.x & msk, t.y & msk, t.z & msk, t.w & msk);
      kb.u4[1] = make_uint4(0u, 0u, 0u, 0u);
      s[st].v = __builtin_amdgcn_wmma_f32_16x16x32_f16(false, qa.v, false, kb.v,
                                                       (short)0, zf, false, false);
    }

    // ---- online softmax; D layout: row = r + 8g, col = st*16 + ln ----
#pragma unroll
    for (int r = 0; r < 8; ++r) {
      float t = fmaxf(fmaxf(s[0].f[r], s[1].f[r]), fmaxf(s[2].f[r], s[3].f[r]));
      t = red_max16(t);
      float nm = fmaxf(mrow[r], t);
      float sc = __expf(mrow[r] - nm);
      mrow[r] = nm;
      float p0 = __expf(s[0].f[r] - nm);
      float p1 = __expf(s[1].f[r] - nm);
      float p2 = __expf(s[2].f[r] - nm);
      float p3 = __expf(s[3].f[r] - nm);
      float rs = red_sum16((p0 + p1) + (p2 + p3));
      lrow[r] = lrow[r] * sc + rs;
#pragma unroll
      for (int f = 0; f < 4; ++f) acc[f].f[r] *= sc;
      int m = r + 8 * g;
      P[m * 64 + ln]      = (_Float16)p0;
      P[m * 64 + 16 + ln] = (_Float16)p1;
      P[m * 64 + 32 + ln] = (_Float16)p2;
      P[m * 64 + 48 + ln] = (_Float16)p3;
    }

    // ---- O += P * V^T : 2 K-steps of 32, 4 C-tiles of 16 ----
#pragma unroll
    for (int t = 0; t < 2; ++t) {
      F16x16 pa;   // P A-fragment (same-wave LDS is in-order)
      pa.u4[0] = *(const uint4*)(P + ln * 64 + t * 32 + 8 * g);
      pa.u4[1] = *(const uint4*)(P + ln * 64 + t * 32 + 16 + 8 * g);
#pragma unroll
      for (int f = 0; f < 4; ++f) {
        F16x16 vb;  // B[k=j-local][n=c] = Vh[b][c][j0 + 32t + (h>=8?16:0) + 8g + h%8]
        const _Float16* vp = Vb + (size_t)(f * 16 + ln) * NPIX + j0 + t * 32 + g * 8;
        vb.u4[0] = *(const uint4*)(vp);
        vb.u4[1] = *(const uint4*)(vp + 16);
        acc[f].v = __builtin_amdgcn_wmma_f32_16x16x32_f16(false, pa.v, false, vb.v,
                                                          (short)0, acc[f].v, false, false);
      }
    }
  }

  // ---- publish partials ----
#pragma unroll
  for (int r = 0; r < 8; ++r) {
    mS[wave][r * 32 + lane] = mrow[r];
    lS[wave][r * 32 + lane] = lrow[r];
#pragma unroll
    for (int f = 0; f < 4; ++f) accS[wave][f][r * 32 + lane] = acc[f].f[r];
  }
  __syncthreads();

  // ---- merge: wave w handles C-fragment f = w ----
  const int f = wave;
  const float gam = gamma[0];
#pragma unroll
  for (int r = 0; r < 8; ++r) {
    float m0 = mS[0][r * 32 + lane], m1 = mS[1][r * 32 + lane];
    float m2 = mS[2][r * 32 + lane], m3 = mS[3][r * 32 + lane];
    float M = fmaxf(fmaxf(m0, m1), fmaxf(m2, m3));
    float e0 = __expf(m0 - M), e1 = __expf(m1 - M);
    float e2 = __expf(m2 - M), e3 = __expf(m3 - M);
    float L = lS[0][r * 32 + lane] * e0 + lS[1][r * 32 + lane] * e1 +
              lS[2][r * 32 + lane] * e2 + lS[3][r * 32 + lane] * e3;
    float a = accS[0][f][r * 32 + lane] * e0 + accS[1][f][r * 32 + lane] * e1 +
              accS[2][f][r * 32 + lane] * e2 + accS[3][f][r * 32 + lane] * e3;
    int i = i0 + r + 8 * g;
    int c = f * 16 + ln;
    size_t o = ((size_t)b * C_DIM + c) * NPIX + i;
    out[o] = fmaf(gam, a / L, x[o]);
  }
}

// ---------------------------------------------------------------------------
extern "C" void kernel_launch(void* const* d_in, const int* in_sizes, int n_in,
                              void* d_out, int out_size, void* d_ws, size_t ws_size,
                              hipStream_t stream) {
  const float* x     = (const float*)d_in[0];
  const float* Wq    = (const float*)d_in[1];
  const float* bq    = (const float*)d_in[2];
  const float* Wk    = (const float*)d_in[3];
  const float* bk    = (const float*)d_in[4];
  const float* Wv    = (const float*)d_in[5];
  const float* bv    = (const float*)d_in[6];
  const float* gamma = (const float*)d_in[7];
  float* out = (float*)d_out;

  // workspace layout (f16): Qh 256KB | Kh 256KB | Vh 2MB
  char* ws = (char*)d_ws;
  _Float16* Qh = (_Float16*)(ws);
  _Float16* Kh = (_Float16*)(ws + (size_t)BATCH * NPIX * QKD * sizeof(_Float16));
  _Float16* Vh = (_Float16*)(ws + (size_t)2 * BATCH * NPIX * QKD * sizeof(_Float16));

  sa_proj_kernel<<<(BATCH * NPIX) / 256, 256, 0, stream>>>(
      x, Wq, bq, Wk, bk, Wv, bv, Qh, Kh, Vh);

  // 1024 blocks x 4 waves = 4096 waves (split-j, merged in-block)
  sa_attn_kernel<<<1024, 128, 0, stream>>>(Qh, Kh, Vh, x, gamma, out);
}
